// FFNDynamicMemories_66073776881809
// MI455X (gfx1250) — matrix operand order repaired and checked by reading, not compiled, not run on hardware
//
#include <hip/hip_runtime.h>
#include <hip/hip_bf16.h>

// ---------------------------------------------------------------------------
// FFN with dynamic (hypernetwork-generated) low-rank memories, CDNA5/gfx1250.
// - All GEMMs on v_wmma_f32_16x16x32_bf16 (wave32).
// - gen_w2 (bf16, ~101 MB, L2-resident) streamed with the Tensor Data Mover:
//   one tensor_load_to_lds per 128x64 B-tile (2D D#, LDS padding reproduces
//   the LDT=80 row stride), waited with s_wait_tensorcnt, double-buffered.
// - Base-path GEMMs stage via global_load_async_to_lds_b128 (ASYNCcnt).
// - Per-token low-rank matvecs refactored into two huge-K GEMMs whose A
//   operand is an on-the-fly outer product (x ⊗ hg, h_mem ⊗ hg).
// ---------------------------------------------------------------------------

#define DIM       512
#define HIDDEN    2048
#define D_M       256
#define RLR       64
#define GENH      512
#define NTOK      1024      // B*T = 2*512

#define LDT       80        // LDS row stride (elements), 160B: 32B-aligned rows

typedef __attribute__((ext_vector_type(16))) __bf16 v16bf;
typedef __attribute__((ext_vector_type(8)))  float  v8f;
typedef __attribute__((ext_vector_type(4)))  unsigned int v4u;
typedef __attribute__((ext_vector_type(8)))  int    v8i;
typedef __attribute__((ext_vector_type(4)))  int    v4i;

#if defined(__has_builtin)
#if __has_builtin(__builtin_amdgcn_tensor_load_to_lds)
#define HAVE_TDM 1
#endif
#endif
#ifndef HAVE_TDM
#define HAVE_TDM 0
#endif

__device__ __forceinline__ unsigned short f2bf(float f) {
    unsigned int u = __float_as_uint(f);
    unsigned int r = u + 0x7fffu + ((u >> 16) & 1u);   // round-to-nearest-even
    return (unsigned short)(r >> 16);
}
__device__ __forceinline__ float bf2f(unsigned short h) {
    return __uint_as_float(((unsigned int)h) << 16);
}
__device__ __forceinline__ float silu_f(float v) {
    return v / (1.0f + __expf(-v));
}

// --- CDNA5 async memory->LDS (ASYNCcnt), no VGPR round trip ----------------
__device__ __forceinline__ void async_copy_b128(unsigned lds_addr, const void* gptr) {
    asm volatile("global_load_async_to_lds_b128 %0, %1, off"
                 :: "v"(lds_addr), "v"((unsigned long long)(size_t)gptr)
                 : "memory");
}
__device__ __forceinline__ void async_wait_all() {
    asm volatile("s_wait_asynccnt 0x0" ::: "memory");
}
// LDS byte address = low 32 bits of the generic pointer (ISA: LDS aperture).
__device__ __forceinline__ unsigned lds_off(const void* p) {
    return (unsigned)(size_t)p;
}

// --- CDNA5 Tensor Data Mover: 2D tile load, bf16 elements ------------------
// D# per CDNA5 ISA. One wave issues; TENSORcnt tracks.
// LDS padding: pad_interval code 4 (=32 DWORDs = 64 bf16) + pad_amount code 7
// (=8 DWORDs = 16 bf16) lands consecutive rows at the LDT=80 element stride.
// This toolchain's builtin is the 6-arg form:
//   (uint32x4 g0, int32x8 g1, int32x4 g2, int32x4 g3, int32x8 gx, i32 cpol)
__device__ __forceinline__ void tdm_load_tile_2d(
    unsigned lds_addr, const void* gptr,
    unsigned tileW, unsigned tileH,                 // elements / rows
    unsigned long long rowStrideElems)
{
#if HAVE_TDM
    const unsigned long long ga = (unsigned long long)(size_t)gptr;
    v4u g0;
    g0[0] = 1u;                                          // count=1 (user D#)
    g0[1] = lds_addr;                                    // lds_addr [63:32]
    g0[2] = (unsigned)(ga & 0xFFFFFFFFu);                // global_addr lo
    g0[3] = (unsigned)((ga >> 32) & 0x01FFFFFFu)         // global_addr hi
          | (2u << 30);                                  // type=2 ("image")
    v8i g1;
    g1[0] = (int)((1u << 16)        // data_size = 1 -> 2 bytes
                | (1u << 20)        // pad_enable
                | (4u << 22)        // pad_interval: 32 DWORDs
                | (7u << 25));      // pad_amount : 8 DWORDs
    g1[1] = (int)((tileW & 0xFFFFu) << 16);              // tensor_dim0 lo16
    g1[2] = (int)(((tileW >> 16) & 0xFFFFu)              // tensor_dim0 hi16
                | ((tileH & 0xFFFFu) << 16));            // tensor_dim1 lo16
    g1[3] = (int)(((tileH >> 16) & 0xFFFFu)              // tensor_dim1 hi16
                | ((tileW & 0xFFFFu) << 16));            // tile_dim0
    g1[4] = (int)(tileH & 0xFFFFu);                      // tile_dim1 (dim2=0)
    g1[5] = (int)(rowStrideElems & 0xFFFFFFFFu);         // dim0_stride lo32
    g1[6] = (int)((rowStrideElems >> 32) & 0xFFFFu);     // dim0_stride hi16
    g1[7] = 0;                                           // dim1_stride hi
    v4i g2 = {0, 0, 0, 0};
    v4i g3 = {0, 0, 0, 0};
    v8i gx = {0, 0, 0, 0, 0, 0, 0, 0};
    __builtin_amdgcn_tensor_load_to_lds(g0, g1, g2, g3, gx, 0);
#else
    (void)lds_addr; (void)gptr; (void)tileW; (void)tileH; (void)rowStrideElems;
#endif
}

// ---------------------------------------------------------------------------
// Core: one 128x128 block tile, K-step 64, 8 waves (2 Mwaves x 4 Nwaves),
// each wave owns a 64x32 output tile = 4x2 WMMA fragments.
// Fragment loads: A (16x32) and B (32x16) both take contiguous k = 16*(lane/16) + s per lane.
// ---------------------------------------------------------------------------
__device__ __forceinline__ void wmma_block(
    const unsigned short* __restrict__ lsA,
    const unsigned short* __restrict__ lsB,
    int waveM, int waveN, int lane, v8f acc[4][2])
{
    const int m  = lane & 15;
    const int hi = lane >> 4;
#pragma unroll
    for (int kk = 0; kk < 64; kk += 32) {
        v16bf a[4];
#pragma unroll
        for (int mi = 0; mi < 4; ++mi) {
            const __bf16* pa = (const __bf16*)(lsA + (waveM*64 + mi*16 + m)*LDT + kk + hi*16);
            v16bf av;
#pragma unroll
            for (int j = 0; j < 16; ++j) { av[j] = pa[j]; }
            a[mi] = av;
        }
#pragma unroll
        for (int ni = 0; ni < 2; ++ni) {
            const v16bf bv = *(const v16bf*)(lsB + (waveN*32 + ni*16 + m)*LDT + kk + hi*16);
#pragma unroll
            for (int mi = 0; mi < 4; ++mi) {
                acc[mi][ni] = __builtin_amdgcn_wmma_f32_16x16x32_bf16(
                    false, a[mi], false, bv, (short)0, acc[mi][ni], false, false);
            }
        }
    }
}

// ---------------------------------------------------------------------------
// Generic GEMM: C[M,N] = A[M,K](bf16,row) * B[N,K](bf16,row)^T with epilogues.
// epi 0: C = acc (+bias[col]) (+C if accum)         -> f32
// epi 1: C = silu(acc + bias[col])                  -> f32
// epi 2: outb = bf16( silu(aux+bias) * (acc+bias2) ) (SwiGLU fusion)
// Double-buffered LDS; tiles staged with async-to-LDS (ASYNCcnt).
// ---------------------------------------------------------------------------
__global__ void __launch_bounds__(256) gemm_bf16_epi(
    const unsigned short* __restrict__ A, int lda,
    const unsigned short* __restrict__ B, int ldb,
    int N, int K,
    float* __restrict__ C, int ldc,
    const float* __restrict__ bias,
    const float* __restrict__ bias2,
    const float* __restrict__ aux, int ldaux,
    unsigned short* __restrict__ outb, int ldob,
    int epi, int accum)
{
    __shared__ __align__(32) unsigned short lsA[2][128 * LDT];
    __shared__ __align__(32) unsigned short lsB[2][128 * LDT];
    const int tid    = threadIdx.x;
    const int lane   = tid & 31;
    const int waveId = tid >> 5;
    const int waveM  = waveId >> 2, waveN = waveId & 3;
    const int rowBase = blockIdx.y * 128;
    const int colBase = blockIdx.x * 128;
    const int sr = tid >> 1;           // staging row 0..127 (2 threads/row)
    const int sh = (tid & 1) * 32;     // staging half (32 elems = 64B)
    const int nt = K >> 6;

    v8f acc[4][2] = {};

    auto stage = [&](int t, int buf) {
        const int k0 = t << 6;
        { // A
            const unsigned short* g = A + (size_t)(rowBase + sr) * lda + k0 + sh;
            const unsigned la = lds_off(&lsA[buf][sr*LDT + sh]);
#pragma unroll
            for (int v = 0; v < 4; ++v)
                async_copy_b128(la + v*16, g + v*8);
        }
        { // B (weight rows == output columns)
            const int n = colBase + sr;
            if (n < N) {
                const unsigned short* g = B + (size_t)n * ldb + k0 + sh;
                const unsigned lb = lds_off(&lsB[buf][sr*LDT + sh]);
#pragma unroll
                for (int v = 0; v < 4; ++v)
                    async_copy_b128(lb + v*16, g + v*8);
                __builtin_prefetch(g + 128, 0, 0);
            } else {
                uint4 z = make_uint4(0u, 0u, 0u, 0u);
#pragma unroll
                for (int v = 0; v < 4; ++v)
                    *(uint4*)&lsB[buf][sr*LDT + sh + v*8] = z;
            }
        }
    };

    stage(0, 0);
    for (int t = 0; t < nt; ++t) {
        const int pp = t & 1;
        async_wait_all();
        __syncthreads();
        if (t + 1 < nt) stage(t + 1, pp ^ 1);
        wmma_block(lsA[pp], lsB[pp], waveM, waveN, lane, acc);
    }

    const int m = lane & 15, hi = lane >> 4;
#pragma unroll
    for (int mi = 0; mi < 4; ++mi) {
#pragma unroll
        for (int ni = 0; ni < 2; ++ni) {
            const int col = colBase + waveN*32 + ni*16 + m;
            if (col >= N) continue;
            const int rbase = rowBase + waveM*64 + mi*16 + hi*8;
#pragma unroll
            for (int j = 0; j < 8; ++j) {
                const int row = rbase + j;
                float v = acc[mi][ni][j];
                if (epi == 0) {
                    if (bias) v += bias[col];
                    const size_t idx = (size_t)row * ldc + col;
                    if (accum) v += C[idx];
                    C[idx] = v;
                } else if (epi == 1) {
                    if (bias) v += bias[col];
                    C[(size_t)row * ldc + col] = silu_f(v);
                } else {
                    float x1v = aux[(size_t)row * ldaux + col];
                    if (bias)  x1v += bias[col];
                    if (bias2) v   += bias2[col];
                    outb[(size_t)row * ldob + col] = f2bf(silu_f(x1v) * v);
                }
            }
        }
    }
}

// ---------------------------------------------------------------------------
// Memory-path GEMM #1: U[1024,128] (u1 || u2) with
//   A[t, k=d*512+h] = x[t,d] * hg[t,h]   (computed into LDS, bf16)
//   B[n, k]         = gen_w2 regions 1&2, one TDM 2D tile per K-step
//                     (row stride = DIM*GENH elements between weight rows)
// K = 262144, split over gridDim.z; results combined with f32 global atomics.
// ---------------------------------------------------------------------------
__global__ void __launch_bounds__(256) mem_u_gemm(
    const float* __restrict__ x,            // [1024,512] f32
    const float* __restrict__ hg,           // [1024,512] f32
    const unsigned short* __restrict__ gw2b,// bf16 [3*R*DIM*GENH]
    float* __restrict__ U,                  // [1024,128], pre-zeroed
    int tilesPer)
{
    __shared__ __align__(32) unsigned short lsA[2][128 * LDT];
    __shared__ __align__(32) unsigned short lsB[2][128 * LDT];
    const int tid = threadIdx.x, lane = tid & 31, waveId = tid >> 5;
    const int waveM = waveId >> 2, waveN = waveId & 3;
    const int rowBase = blockIdx.y * 128;
    const int sr = tid >> 1, sh = (tid & 1) * 32;
    const int t0 = blockIdx.z * tilesPer;

    v8f acc[4][2] = {};

    auto stage = [&](int tile, int buf) {
        const int k0 = tile * 64;
        const int d  = k0 >> 9;        // k = d*512 + h
        const int h0 = k0 & 511;
        // B tile: one TDM op covers all 128 rows (issued by wave 0)
        if (waveId == 0) {
#if HAVE_TDM
            tdm_load_tile_2d(lds_off(&lsB[buf][0]),
                             gw2b + (size_t)d * GENH + h0,
                             64, 128, (unsigned long long)DIM * GENH);
#else
            ;
#endif
        }
#if !HAVE_TDM
        {
            const unsigned short* g = gw2b + ((size_t)sr * DIM + d) * GENH + h0 + sh;
            const unsigned lb = lds_off(&lsB[buf][sr*LDT + sh]);
#pragma unroll
            for (int v = 0; v < 4; ++v)
                async_copy_b128(lb + v*16, g + v*8);
        }
#endif
        { // A tile: outer product x[t,d] * hg[t, h0+sh .. +31] (vectorized)
            const int t = rowBase + sr;
            const float xv = x[(size_t)t * DIM + d];
            const float4* hp = (const float4*)(hg + (size_t)t * GENH + h0 + sh);
            ushort4* dst = (ushort4*)&lsA[buf][sr*LDT + sh];
#pragma unroll
            for (int j = 0; j < 8; ++j) {
                float4 f = hp[j];
                dst[j] = make_ushort4(f2bf(xv * f.x), f2bf(xv * f.y),
                                      f2bf(xv * f.z), f2bf(xv * f.w));
            }
        }
    };

    stage(t0, 0);
    for (int tile = t0; tile < t0 + tilesPer; ++tile) {
        const int pp = (tile - t0) & 1;
#if HAVE_TDM
        if (waveId == 0) __builtin_amdgcn_s_wait_tensorcnt(0);
#else
        async_wait_all();
#endif
        __syncthreads();
        if (tile + 1 < t0 + tilesPer) stage(tile + 1, pp ^ 1);
        wmma_block(lsA[pp], lsB[pp], waveM, waveN, lane, acc);
    }

    const int m = lane & 15, hi = lane >> 4;
#pragma unroll
    for (int mi = 0; mi < 4; ++mi)
#pragma unroll
        for (int ni = 0; ni < 2; ++ni) {
            const int col = waveN*32 + ni*16 + m;          // N = 128 exactly
#pragma unroll
            for (int j = 0; j < 8; ++j) {
                const int row = rowBase + waveM*64 + mi*16 + hi*8 + j;
                __hip_atomic_fetch_add(&U[(size_t)row * 128 + col],
                                       acc[mi][ni][j],
                                       __ATOMIC_RELAXED, __HIP_MEMORY_SCOPE_AGENT);
            }
        }
}

// ---------------------------------------------------------------------------
// Memory-path GEMM #2: y_mem[1024,512] with
//   A[t, k=r*512+h] = h_mem[t,r] * hg[t,h]  (computed into LDS)
//   B[d, k]         = gen_w2 region 3 viewed [512, 32768] row-major,
//                     one TDM 2D tile per K-step (row stride = R*GENH)
// K = 32768, split over gridDim.z, f32 atomics into pre-zeroed y_mem.
// ---------------------------------------------------------------------------
__global__ void __launch_bounds__(256) mem_y_gemm(
    const unsigned short* __restrict__ hm,  // bf16 [1024,64]
    const float* __restrict__ hg,           // [1024,512] f32
    const unsigned short* __restrict__ gw2b,
    float* __restrict__ ymem,               // [1024,512], pre-zeroed
    int tilesPer)
{
    __shared__ __align__(32) unsigned short lsA[2][128 * LDT];
    __shared__ __align__(32) unsigned short lsB[2][128 * LDT];
    const int tid = threadIdx.x, lane = tid & 31, waveId = tid >> 5;
    const int waveM = waveId >> 2, waveN = waveId & 3;
    const int rowBase = blockIdx.y * 128;
    const int colBase = blockIdx.x * 128;
    const int sr = tid >> 1, sh = (tid & 1) * 32;
    const int t0 = blockIdx.z * tilesPer;
    const size_t G3OFF = (size_t)2 * RLR * DIM * GENH;   // region-3 base

    v8f acc[4][2] = {};

    auto stage = [&](int tile, int buf) {
        const int k0 = tile * 64;
        const int r  = k0 >> 9;        // k = r*512 + h
        const int h0 = k0 & 511;
        if (waveId == 0) {
#if HAVE_TDM
            tdm_load_tile_2d(lds_off(&lsB[buf][0]),
                             gw2b + G3OFF + (size_t)colBase * (RLR * GENH) + k0,
                             64, 128, (unsigned long long)RLR * GENH);
#else
            ;
#endif
        }
#if !HAVE_TDM
        {
            const unsigned short* g =
                gw2b + G3OFF + (size_t)(colBase + sr) * (RLR * GENH) + k0 + sh;
            const unsigned lb = lds_off(&lsB[buf][sr*LDT + sh]);
#pragma unroll
            for (int v = 0; v < 4; ++v)
                async_copy_b128(lb + v*16, g + v*8);
        }
#endif
        { // A tile: h_mem[t,r] * hg[t, h0+sh..]
            const int t = rowBase + sr;
            const float hv = bf2f(hm[(size_t)t * RLR + r]);
            const float4* hp = (const float4*)(hg + (size_t)t * GENH + h0 + sh);
            ushort4* dst = (ushort4*)&lsA[buf][sr*LDT + sh];
#pragma unroll
            for (int j = 0; j < 8; ++j) {
                float4 f = hp[j];
                dst[j] = make_ushort4(f2bf(hv * f.x), f2bf(hv * f.y),
                                      f2bf(hv * f.z), f2bf(hv * f.w));
            }
        }
    };

    stage(t0, 0);
    for (int tile = t0; tile < t0 + tilesPer; ++tile) {
        const int pp = (tile - t0) & 1;
#if HAVE_TDM
        if (waveId == 0) __builtin_amdgcn_s_wait_tensorcnt(0);
#else
        async_wait_all();
#endif
        __syncthreads();
        if (tile + 1 < t0 + tilesPer) stage(tile + 1, pp ^ 1);
        wmma_block(lsA[pp], lsB[pp], waveM, waveN, lane, acc);
    }

    const int m = lane & 15, hi = lane >> 4;
#pragma unroll
    for (int mi = 0; mi < 4; ++mi)
#pragma unroll
        for (int ni = 0; ni < 2; ++ni) {
            const int col = colBase + waveN*32 + ni*16 + m;
#pragma unroll
            for (int j = 0; j < 8; ++j) {
                const int row = rowBase + waveM*64 + mi*16 + hi*8 + j;
                __hip_atomic_fetch_add(&ymem[(size_t)row * DIM + col],
                                       acc[mi][ni][j],
                                       __ATOMIC_RELAXED, __HIP_MEMORY_SCOPE_AGENT);
            }
        }
}

// ------------------------- small elementwise kernels -----------------------
__global__ void cvt_f32_bf16x4(const float* __restrict__ s,
                               unsigned short* __restrict__ d, int n4)
{
    int i = blockIdx.x * blockDim.x + threadIdx.x;
    const int stride = gridDim.x * blockDim.x;
    for (; i < n4; i += stride) {
        float4 f = ((const float4*)s)[i];
        ((ushort4*)d)[i] = make_ushort4(f2bf(f.x), f2bf(f.y), f2bf(f.z), f2bf(f.w));
    }
}

__global__ void zero_f32(float* __restrict__ p, int n)
{
    int i = blockIdx.x * blockDim.x + threadIdx.x;
    const int stride = gridDim.x * blockDim.x;
    for (; i < n; i += stride) p[i] = 0.0f;
}

__global__ void hmem_silu_kernel(const float* __restrict__ U,
                                 unsigned short* __restrict__ hm, int n)
{
    int i = blockIdx.x * blockDim.x + threadIdx.x;
    const int stride = gridDim.x * blockDim.x;
    for (; i < n; i += stride) {
        const int t = i >> 6, r = i & 63;
        const float u1 = U[(size_t)t * 128 + r];
        const float u2 = U[(size_t)t * 128 + 64 + r];
        hm[i] = f2bf(silu_f(u1) * u2);
    }
}

__global__ void final_add_kernel(float* __restrict__ out,
                                 const float* __restrict__ ymem,
                                 const float* __restrict__ mg, int n)
{
    const float g = 1.0f / (1.0f + __expf(-mg[0]));   // sigmoid(mem_gate)
    int i = blockIdx.x * blockDim.x + threadIdx.x;
    const int stride = gridDim.x * blockDim.x;
    for (; i < n; i += stride) out[i] += g * ymem[i];
}

// ---------------------------------------------------------------------------
extern "C" void kernel_launch(void* const* d_in, const int* in_sizes, int n_in,
                              void* d_out, int out_size, void* d_ws, size_t ws_size,
                              hipStream_t stream)
{
    (void)in_sizes; (void)n_in; (void)out_size; (void)ws_size;

    const float* x      = (const float*)d_in[0];   // [1024,512]
    const float* m_tok  = (const float*)d_in[1];   // [1024,256]
    const float* W1     = (const float*)d_in[2];   // [2048,512]
    const float* W2     = (const float*)d_in[3];   // [2048,512]
    const float* W3     = (const float*)d_in[4];   // [512,2048]
    const float* b1     = (const float*)d_in[5];
    const float* b2     = (const float*)d_in[6];
    const float* b3     = (const float*)d_in[7];
    const float* gen_w1 = (const float*)d_in[8];   // [512,256]
    const float* gen_b1 = (const float*)d_in[9];
    const float* gen_w2 = (const float*)d_in[10];  // [98304,512]
    const float* gen_b2 = (const float*)d_in[11];  // [98304]
    const float* mgate  = (const float*)d_in[12];
    float* out = (float*)d_out;                    // [1024,512] f32

    // workspace carve-out (256B aligned slots)
    char* wsb = (char*)d_ws;
    size_t o = 0;
    auto carve = [&](size_t bytes) -> char* {
        char* p = wsb + o;
        o += (bytes + 255) & ~(size_t)255;
        return p;
    };
    unsigned short* xb    = (unsigned short*)carve((size_t)NTOK * DIM * 2);
    unsigned short* mtb   = (unsigned short*)carve((size_t)NTOK * D_M * 2);
    unsigned short* W1b   = (unsigned short*)carve((size_t)HIDDEN * DIM * 2);
    unsigned short* W2b   = (unsigned short*)carve((size_t)HIDDEN * DIM * 2);
    unsigned short* W3b   = (unsigned short*)carve((size_t)DIM * HIDDEN * 2);
    unsigned short* gw1b  = (unsigned short*)carve((size_t)GENH * D_M * 2);
    unsigned short* gw2b  = (unsigned short*)carve((size_t)3 * RLR * DIM * GENH * 2); // ~101 MB, L2-resident
    unsigned short* gb2b  = (unsigned short*)carve((size_t)3 * RLR * DIM * 2);
    float*          hgf   = (float*)carve((size_t)NTOK * GENH * 4);
    float*          x1f   = (float*)carve((size_t)NTOK * HIDDEN * 4);
    unsigned short* hbase = (unsigned short*)carve((size_t)NTOK * HIDDEN * 2);
    float*          U     = (float*)carve((size_t)NTOK * 128 * 4);
    unsigned short* hmb   = (unsigned short*)carve((size_t)NTOK * RLR * 2);
    float*          ymem  = (float*)carve((size_t)NTOK * DIM * 4);

    auto cvt = [&](const float* s, unsigned short* d, int n) {
        const int n4 = n / 4;
        int blocks = (n4 + 255) / 256; if (blocks > 4096) blocks = 4096;
        cvt_f32_bf16x4<<<blocks, 256, 0, stream>>>(s, d, n4);
    };
    cvt(x,      xb,   NTOK * DIM);
    cvt(m_tok,  mtb,  NTOK * D_M);
    cvt(W1,     W1b,  HIDDEN * DIM);
    cvt(W2,     W2b,  HIDDEN * DIM);
    cvt(W3,     W3b,  DIM * HIDDEN);
    cvt(gen_w1, gw1b, GENH * D_M);
    cvt(gen_w2, gw2b, 3 * RLR * DIM * GENH);
    cvt(gen_b2, gb2b, 3 * RLR * DIM);

    zero_f32<<<256, 256, 0, stream>>>(U,    NTOK * 128);
    zero_f32<<<512, 256, 0, stream>>>(ymem, NTOK * DIM);

    // hg = silu(m_tok @ gen_w1^T + gen_b1)   [1024,512] f32
    gemm_bf16_epi<<<dim3(GENH/128, NTOK/128), 256, 0, stream>>>(
        mtb, D_M, gw1b, D_M, GENH, D_M, hgf, GENH,
        gen_b1, nullptr, nullptr, 0, nullptr, 0, /*epi=*/1, 0);

    // x1 = x @ W1^T   [1024,2048] f32 (bias folded into SwiGLU epilogue)
    gemm_bf16_epi<<<dim3(HIDDEN/128, NTOK/128), 256, 0, stream>>>(
        xb, DIM, W1b, DIM, HIDDEN, DIM, x1f, HIDDEN,
        nullptr, nullptr, nullptr, 0, nullptr, 0, /*epi=*/0, 0);

    // h_base = silu(x1+b1) * (x@W2^T + b2)  -> bf16 [1024,2048]
    gemm_bf16_epi<<<dim3(HIDDEN/128, NTOK/128), 256, 0, stream>>>(
        xb, DIM, W2b, DIM, HIDDEN, DIM, x1f, HIDDEN,
        b1, b2, x1f, HIDDEN, hbase, HIDDEN, /*epi=*/2, 0);

    // U = [u1 || u2]: K = 262144 outer-product GEMM, 16-way K-split (TDM)
    mem_u_gemm<<<dim3(1, NTOK/128, 16), 256, 0, stream>>>(
        x, hgf, gw2b, U, /*tilesPer=*/(DIM * GENH / 64) / 16);

    // + gen_b2 contributions: u1 += x@gb2_1^T ; u2 += x@gb2_2^T
    gemm_bf16_epi<<<dim3(1, NTOK/128), 256, 0, stream>>>(
        xb, DIM, gb2b, DIM, RLR, DIM, U, 128,
        nullptr, nullptr, nullptr, 0, nullptr, 0, 0, /*accum=*/1);
    gemm_bf16_epi<<<dim3(1, NTOK/128), 256, 0, stream>>>(
        xb, DIM, gb2b + (size_t)RLR * DIM, DIM, RLR, DIM, U + 64, 128,
        nullptr, nullptr, nullptr, 0, nullptr, 0, 0, /*accum=*/1);

    // h_mem = silu(u1) * u2  -> bf16 [1024,64]
    hmem_silu_kernel<<<256, 256, 0, stream>>>(U, hmb, NTOK * RLR);

    // y_mem: K = 32768 outer-product GEMM, 4-way K-split (TDM)
    mem_y_gemm<<<dim3(DIM/128, NTOK/128, 4), 256, 0, stream>>>(
        hmb, hgf, gw2b, ymem, /*tilesPer=*/(RLR * GENH / 64) / 4);

    // + gen_b2 region-3 bias: y_mem += h_mem @ gb2_3^T  (B viewed [512,64])
    gemm_bf16_epi<<<dim3(DIM/128, NTOK/128), 256, 0, stream>>>(
        hmb, RLR, gb2b + (size_t)2 * RLR * DIM, RLR, DIM, RLR, ymem, DIM,
        nullptr, nullptr, nullptr, 0, nullptr, 0, 0, /*accum=*/1);

    // y_base = h_base @ W3^T + b3  -> out
    gemm_bf16_epi<<<dim3(DIM/128, NTOK/128), 256, 0, stream>>>(
        hbase, HIDDEN, W3b, HIDDEN, DIM, HIDDEN, out, DIM,
        b3, nullptr, nullptr, 0, nullptr, 0, 0, 0);

    // out += sigmoid(mem_gate) * y_mem
    final_add_kernel<<<512, 256, 0, stream>>>(out, ymem, mgate, NTOK * DIM);
}
